// TJSCC_PCT_36661840839346
// MI455X (gfx1250) — matrix operand order repaired and checked by reading, not compiled, not run on hardware
//
#include <hip/hip_runtime.h>
#include <math.h>

#define BATCH 4
#define NPTS  4096
#define KNNK  32
#define NSAMP 32

typedef __attribute__((ext_vector_type(16))) _Float16 v16h;
typedef __attribute__((ext_vector_type(8)))  float    v8f;

// ---- gfx1250 async-copy / wait helpers (guarded; asm fallback) ----
#if defined(__has_builtin) && __has_builtin(__builtin_amdgcn_s_wait_asynccnt)
#define ASYNC_WAIT0() __builtin_amdgcn_s_wait_asynccnt(0)
#else
#define ASYNC_WAIT0() asm volatile("s_wait_asynccnt 0" ::: "memory")
#endif
#if defined(__has_builtin) && __has_builtin(__builtin_amdgcn_s_wait_tensorcnt)
#define TENSOR_WAIT0() __builtin_amdgcn_s_wait_tensorcnt(0)
#else
#define TENSOR_WAIT0() asm volatile("s_wait_tensorcnt 0" ::: "memory")
#endif

// Issue one 16B global->LDS async copy. lds = LDS byte offset (low 32 bits of
// a generic pointer to __shared__), ga = 64-bit global address.
__device__ __forceinline__ void async_g2l_b128(unsigned lds, unsigned long long ga)
{
  asm volatile("global_load_async_to_lds_b128 %0, %1, off"
               :: "v"(lds), "v"(ga) : "memory");
}

// =====================================================================
// WMMA GEMM:  C[M,N] = A[M,K] @ W[K,N] (+bias) (+res) (+relu)
// f32 in/out, f16 WMMA compute with f32 accumulate.
// Block = 128 threads = 4 waves; each wave owns a 16x16 C tile;
// block covers 64(M) x 16(N); K-loop step 32.
// Tiles staged in LDS as raw f32 (async global->LDS fast path),
// converted to f16 per-lane at fragment build.
// =====================================================================
#define GTM 64
#define GTN 16
#define GTK 32

__global__ __launch_bounds__(128)
void k_gemm(const float* __restrict__ A, const float* __restrict__ W,
            const float* __restrict__ bias, const float* __restrict__ res,
            float* __restrict__ C, int M, int K, int N, int relu)
{
  __shared__ float As32[GTM][GTK];   // 8KB, row = 128B
  __shared__ float Ws32[GTK][GTN];   // 2KB, row = 64B
  const int tid  = threadIdx.x;
  const int wave = tid >> 5;
  const int lane = tid & 31;
  const int half = lane >> 4;    // 0: lanes 0-15, 1: lanes 16-31
  const int ml   = lane & 15;
  const int m0 = blockIdx.x * GTM;
  const int n0 = blockIdx.y * GTN;
  const int arow = wave * 16 + ml;

  const bool fastA = (m0 + GTM <= M) && ((K & 3) == 0);
  const bool fastW = (n0 + GTN <= N) && ((N & 3) == 0);

  v8f acc = {};
  for (int k0 = 0; k0 < K; k0 += GTK) {
    const bool kin = (k0 + GTK <= K);
    // ---- stage A tile ----
    if (fastA && kin) {
      // 64x32 f32 = 512 x 16B chunks, 4 per thread, async direct to LDS
      #pragma unroll
      for (int u = 0; u < 4; ++u) {
        int chunk = tid + u * 128;
        int r = chunk >> 3, c4 = (chunk & 7) << 2;
        unsigned lds = (unsigned)(size_t)&As32[r][c4];
        unsigned long long ga =
            (unsigned long long)(size_t)(A + (size_t)(m0 + r) * K + k0 + c4);
        async_g2l_b128(lds, ga);
      }
    } else {
      for (int i = tid; i < GTM * GTK; i += 128) {
        int r = i >> 5, cc = i & 31;
        int gr = m0 + r, gc = k0 + cc;
        int cr = gr < M ? gr : M - 1;          // clamp, then select-zero:
        int ccl = gc < K ? gc : K - 1;         // avoids divergent load guards
        float v = A[(size_t)cr * K + ccl];
        As32[r][cc] = (gr < M && gc < K) ? v : 0.f;
      }
    }
    // ---- stage W tile ----
    if (fastW && kin) {
      // 32x16 f32 = 128 x 16B chunks, 1 per thread
      int r = tid >> 2, c4 = (tid & 3) << 2;
      unsigned lds = (unsigned)(size_t)&Ws32[r][c4];
      unsigned long long ga =
          (unsigned long long)(size_t)(W + (size_t)(k0 + r) * N + n0 + c4);
      async_g2l_b128(lds, ga);
    } else {
      for (int i = tid; i < GTK * GTN; i += 128) {
        int kk = i >> 4, nn = i & 15;
        int gk = k0 + kk, gn = n0 + nn;
        int ck = gk < K ? gk : K - 1;
        int cn = gn < N ? gn : N - 1;
        float v = W[(size_t)ck * N + cn];
        Ws32[kk][nn] = (gk < K && gn < N) ? v : 0.f;
      }
    }
    if ((fastA && kin) || (fastW && kin)) ASYNC_WAIT0();
    __syncthreads();
    // Prefetch next A K-tile rows (lowers to global_prefetch_b8)
    if (k0 + GTK < K) {
      int pr = m0 + (tid >> 1);
      if (pr < M) __builtin_prefetch(&A[(size_t)pr * K + k0 + GTK], 0, 3);
    }
    // Build fragments per CDNA5 ISA VGPR layouts (f32 LDS -> f16 regs).
    // A 16x32 f16: lanes0-15 K={0..7,16..23}, lanes16-31 K={8..15,24..31}
    v16h a, b;
    #pragma unroll
    for (int j = 0; j < 8; ++j) {
      a[j]     = (_Float16)As32[arow][half * 8 + j];
      a[8 + j] = (_Float16)As32[arow][16 + half * 8 + j];
    }
    // B 32x16 f16: lanes0-15 K={0..7,16..23}, lanes16-31 K={8..15,24..31}, N=lane&15
    #pragma unroll
    for (int j = 0; j < 16; ++j) b[j] = (_Float16)Ws32[(j < 8 ? half * 8 + j : 16 + half * 8 + (j - 8))][ml];
    acc = __builtin_amdgcn_wmma_f32_16x16x32_f16(false, a, false, b,
                                                 (short)0, acc, false, false);
    __syncthreads();
  }
  TENSOR_WAIT0();   // tensor-counter path (no-op here; TENSORcnt==0)
  // C f32 16x16: VGPR r -> M = r + half*8 ; N = lane&15
  #pragma unroll
  for (int r = 0; r < 8; ++r) {
    int gm = m0 + wave * 16 + half * 8 + r;
    int gn = n0 + ml;
    if (gm < M && gn < N) {
      float v = acc[r];
      if (bias) v += bias[gn];
      if (res)  v += res[(size_t)gm * N + gn];
      if (relu) v = fmaxf(v, 0.f);
      C[(size_t)gm * N + gn] = v;
    }
  }
}

// =====================================================================
// BatchNorm (population mean/var over rows) + ReLU
// =====================================================================
__global__ void k_bn_stats(const float* __restrict__ X, int M, int C,
                           float* __restrict__ mv)
{
  int c = blockIdx.x;
  __shared__ float s1[256], s2[256];
  float a = 0.f, b = 0.f;
  for (int i = threadIdx.x; i < M; i += 256) {
    float v = X[(size_t)i * C + c];
    a += v; b += v * v;
  }
  s1[threadIdx.x] = a; s2[threadIdx.x] = b;
  __syncthreads();
  for (int o = 128; o > 0; o >>= 1) {
    if (threadIdx.x < o) { s1[threadIdx.x] += s1[threadIdx.x + o];
                           s2[threadIdx.x] += s2[threadIdx.x + o]; }
    __syncthreads();
  }
  if (threadIdx.x == 0) {
    float m = s1[0] / (float)M;
    mv[c]     = m;
    mv[C + c] = s2[0] / (float)M - m * m;
  }
}

__global__ void k_bn_apply(float* __restrict__ X, const float* __restrict__ mv,
                           int total, int C, int relu)
{
  int t = blockIdx.x * 256 + threadIdx.x;
  if (t >= total) return;
  int c = t % C;
  float v = (X[t] - mv[c]) * rsqrtf(mv[C + c] + 1e-5f);
  if (relu) v = fmaxf(v, 0.f);
  X[t] = v;
}

// =====================================================================
// Farthest point sampling: one block per batch, iterative argmax
// =====================================================================
__global__ void k_fps(const float* __restrict__ xyz, int N, int S,
                      int* __restrict__ out)
{
  int b = blockIdx.x;
  const float* P = xyz + (size_t)b * N * 3;
  __shared__ float dist[NPTS];
  __shared__ float rv[256];
  __shared__ int   ri[256];
  __shared__ int   sfar;
  for (int i = threadIdx.x; i < N; i += 256) dist[i] = 1e10f;
  if (threadIdx.x == 0) sfar = 0;
  __syncthreads();
  for (int s = 0; s < S; ++s) {
    int far = sfar;
    if (threadIdx.x == 0) out[b * S + s] = far;
    float cx = P[far * 3], cy = P[far * 3 + 1], cz = P[far * 3 + 2];
    float bm = -1.f; int bi = 0;
    for (int i = threadIdx.x; i < N; i += 256) {
      float dx = P[i * 3] - cx, dy = P[i * 3 + 1] - cy, dz = P[i * 3 + 2] - cz;
      float d = dx * dx + dy * dy + dz * dz;
      float nd = fminf(dist[i], d);
      dist[i] = nd;
      if (nd > bm) { bm = nd; bi = i; }
    }
    rv[threadIdx.x] = bm; ri[threadIdx.x] = bi;
    __syncthreads();
    for (int o = 128; o > 0; o >>= 1) {
      if (threadIdx.x < o && rv[threadIdx.x + o] > rv[threadIdx.x]) {
        rv[threadIdx.x] = rv[threadIdx.x + o];
        ri[threadIdx.x] = ri[threadIdx.x + o];
      }
      __syncthreads();
    }
    if (threadIdx.x == 0) sfar = ri[0];
    __syncthreads();
  }
}

// out[b,s,c] = src[b, idx[b,s], c]
__global__ void k_gather(const float* __restrict__ src, const int* __restrict__ idx,
                         float* __restrict__ out, int N, int S, int C, int total)
{
  int t = blockIdx.x * 256 + threadIdx.x;
  if (t >= total) return;
  int c = t % C, sp = t / C, b = sp / S;
  out[t] = src[((size_t)b * N + idx[sp]) * C + c];
}

// ball query: ascending-index selection of up to NSAMP pts with d2<=r2
__global__ void k_ball(const float* __restrict__ xyz, const float* __restrict__ nxyz,
                       int* __restrict__ gi, int N, int S, float r2, int total)
{
  int t = blockIdx.x * 256 + threadIdx.x;
  if (t >= total) return;
  int b = t / S;
  const float* P = xyz + (size_t)b * N * 3;
  float qx = nxyz[(size_t)t * 3], qy = nxyz[(size_t)t * 3 + 1], qz = nxyz[(size_t)t * 3 + 2];
  int cnt = 0, first = 0; bool have = false;
  for (int j = 0; j < N && cnt < NSAMP; ++j) {
    float dx = P[j * 3] - qx, dy = P[j * 3 + 1] - qy, dz = P[j * 3 + 2] - qz;
    float d2 = dx * dx + dy * dy + dz * dz;
    if (d2 <= r2) {
      if (!have) { first = j; have = true; }
      gi[(size_t)t * NSAMP + cnt] = j;
      ++cnt;
    }
  }
  for (; cnt < NSAMP; ++cnt) gi[(size_t)t * NSAMP + cnt] = first;
}

// g[b,s,j,:] = concat(xyz[gi]-new_xyz, feats[gi])
__global__ void k_group(const float* __restrict__ xyz, const float* __restrict__ feats,
                        const int* __restrict__ gi, const float* __restrict__ nxyz,
                        float* __restrict__ g, int N, int S, int C, int total)
{
  int t = blockIdx.x * 256 + threadIdx.x;
  if (t >= total) return;
  int sp = t / NSAMP, b = sp / S;
  int src = gi[t];
  float* row = g + (size_t)t * (3 + C);
  const float* ps = xyz + ((size_t)b * N + src) * 3;
  const float* pq = nxyz + (size_t)sp * 3;
  row[0] = ps[0] - pq[0]; row[1] = ps[1] - pq[1]; row[2] = ps[2] - pq[2];
  const float* fs = feats + ((size_t)b * N + src) * C;
  for (int c = 0; c < C; ++c) row[3 + c] = fs[c];
}

// out[b,s,c] = max_j X[b,s,j,c]
__global__ void k_maxpool(const float* __restrict__ X, float* __restrict__ out,
                          int S, int C, int total)
{
  int t = blockIdx.x * 256 + threadIdx.x;
  if (t >= total) return;
  int c = t % C; int sp = t / C;
  const float* base = X + ((size_t)sp * NSAMP) * C + c;
  float m = -1e30f;
  for (int j = 0; j < NSAMP; ++j) m = fmaxf(m, base[(size_t)j * C]);
  out[t] = m;
}

// kNN (k=32) by insertion sort; total = B*N, points per batch = N
__global__ void k_knn(const float* __restrict__ xyz, int* __restrict__ knn,
                      int N, int total)
{
  int t = blockIdx.x * 256 + threadIdx.x;
  if (t >= total) return;
  int b = t / N;
  const float* P = xyz + (size_t)b * N * 3;
  float qx = xyz[(size_t)t * 3], qy = xyz[(size_t)t * 3 + 1], qz = xyz[(size_t)t * 3 + 2];
  float bd[KNNK]; int bj[KNNK];
  for (int k = 0; k < KNNK; ++k) { bd[k] = 1e30f; bj[k] = 0; }
  for (int j = 0; j < N; ++j) {
    float dx = P[j * 3] - qx, dy = P[j * 3 + 1] - qy, dz = P[j * 3 + 2] - qz;
    float d2 = dx * dx + dy * dy + dz * dz;
    if (d2 < bd[KNNK - 1]) {
      int p = KNNK - 1;
      while (p > 0 && bd[p - 1] > d2) { bd[p] = bd[p - 1]; bj[p] = bj[p - 1]; --p; }
      bd[p] = d2; bj[p] = j;
    }
  }
  for (int k = 0; k < KNNK; ++k) knn[(size_t)t * KNNK + k] = bj[k];
}

// pxyz[p,j,:] = xyz[p,:] - xyz[knn[p,j],:]   (per-batch pointers; b = p/S = 0)
__global__ void k_posxyz(const float* __restrict__ xyz, const int* __restrict__ knn,
                         float* __restrict__ pxyz, int S, int total)
{
  int t = blockIdx.x * 256 + threadIdx.x;
  if (t >= total) return;
  int p = t / KNNK;
  int n = knn[t];
  int b = p / S;
  for (int k = 0; k < 3; ++k)
    pxyz[(size_t)t * 3 + k] = xyz[(size_t)p * 3 + k] - xyz[((size_t)b * S + n) * 3 + k];
}

// t[p,j,c] = q[p,c] - kx[knn[p,j],c] + pos[p,j,c]
__global__ void k_build_t(const float* __restrict__ q, const float* __restrict__ kx,
                          const int* __restrict__ knn, const float* __restrict__ pos,
                          float* __restrict__ tt, int S, int d, int total)
{
  int e = blockIdx.x * 256 + threadIdx.x;
  if (e >= total) return;
  int c = e % d; int r = e / d;
  int p = r / KNNK; int b = p / S;
  int n = knn[r];
  tt[e] = q[(size_t)p * d + c] - kx[((size_t)b * S + n) * d + c] + pos[e];
}

// softmax over neighbor axis (32), scaled by 1/sqrt(d); in-place
__global__ void k_softmax(float* __restrict__ attn, int d, float scale, int total)
{
  int t = blockIdx.x * 256 + threadIdx.x;
  if (t >= total) return;
  int c = t % d; int p = t / d;
  float* base = attn + (size_t)p * KNNK * d + c;
  float m = -1e30f;
  for (int j = 0; j < KNNK; ++j) m = fmaxf(m, base[(size_t)j * d] * scale);
  float s = 0.f;
  for (int j = 0; j < KNNK; ++j) s += expf(base[(size_t)j * d] * scale - m);
  float inv = 1.f / s;
  for (int j = 0; j < KNNK; ++j) {
    float v = expf(base[(size_t)j * d] * scale - m) * inv;
    base[(size_t)j * d] = v;
  }
}

// res[p,c] = sum_j attn[p,j,c] * (vx[knn[p,j],c] + pos[p,j,c])
__global__ void k_agg(const float* __restrict__ attn, const float* __restrict__ vx,
                      const int* __restrict__ knn, const float* __restrict__ pos,
                      float* __restrict__ res, int S, int d, int total)
{
  int t = blockIdx.x * 256 + threadIdx.x;
  if (t >= total) return;
  int c = t % d; int p = t / d; int b = p / S;
  const float* ab = attn + (size_t)p * KNNK * d + c;
  const float* pb = pos + (size_t)p * KNNK * d + c;
  const int* kb = knn + (size_t)p * KNNK;
  float acc = 0.f;
  for (int j = 0; j < KNNK; ++j) {
    float v = vx[((size_t)b * S + kb[j]) * d + c] + pb[(size_t)j * d];
    acc += ab[(size_t)j * d] * v;
  }
  res[t] = acc;
}

// latent[b,c] = mean_s X[b,s,c]
__global__ void k_mean1(const float* __restrict__ X, float* __restrict__ out,
                        int S, int C, int total)
{
  int t = blockIdx.x * 256 + threadIdx.x;
  if (t >= total) return;
  int c = t % C, b = t / C;
  float s = 0.f;
  for (int i = 0; i < S; ++i) s += X[((size_t)b * S + i) * C + c];
  out[t] = s / (float)S;
}

// power normalize to sqrt(256)=16 and add AWGN
__global__ void k_powernoise(const float* __restrict__ t, const float* __restrict__ noise,
                             float* __restrict__ y)
{
  int b = blockIdx.x, c = threadIdx.x;  // 256 threads
  __shared__ float red[256];
  float v = t[b * 256 + c];
  red[c] = v * v;
  __syncthreads();
  for (int o = 128; o > 0; o >>= 1) {
    if (c < o) red[c] += red[c + o];
    __syncthreads();
  }
  float nrm = sqrtf(red[0]);
  y[b * 256 + c] = v * 16.f / nrm + 0.31622776601683794f * noise[b * 256 + c];
}

// cat[b,n,c] = c<C1 ? p1 : z[b] (broadcast decode vector)
__global__ void k_concat_b(const float* __restrict__ p1, const float* __restrict__ z,
                           float* __restrict__ cat, int C1, int C2, int S, int total)
{
  int e = blockIdx.x * 256 + threadIdx.x;
  if (e >= total) return;
  int Ct = C1 + C2;
  int c = e % Ct; int t = e / Ct; int b = t / S;
  cat[e] = (c < C1) ? p1[(size_t)t * C1 + c] : z[b * C2 + (c - C1)];
}

// cat[:, :C1] = p1
__global__ void k_copycat(const float* __restrict__ p1, float* __restrict__ cat,
                          int C1, int Ct, int total)
{
  int e = blockIdx.x * 256 + threadIdx.x;
  if (e >= total) return;
  int c = e % C1; int t = e / C1;
  cat[(size_t)t * Ct + c] = p1[e];
}

// 3-NN inverse-distance interpolation into cat[:, C1:]
__global__ void k_interp3(const float* __restrict__ x1, const float* __restrict__ x2,
                          const float* __restrict__ p2, float* __restrict__ cat,
                          int N1, int S2, int C1, int C2, int total)
{
  int t = blockIdx.x * 256 + threadIdx.x;
  if (t >= total) return;
  int b = t / N1;
  const float* X2 = x2 + (size_t)b * S2 * 3;
  float qx = x1[(size_t)t * 3], qy = x1[(size_t)t * 3 + 1], qz = x1[(size_t)t * 3 + 2];
  float d0 = 1e30f, d1 = 1e30f, d2 = 1e30f; int i0 = 0, i1 = 0, i2 = 0;
  for (int j = 0; j < S2; ++j) {
    float dx = X2[j * 3] - qx, dy = X2[j * 3 + 1] - qy, dz = X2[j * 3 + 2] - qz;
    float d = dx * dx + dy * dy + dz * dz;
    if (d < d0)      { d2 = d1; i2 = i1; d1 = d0; i1 = i0; d0 = d; i0 = j; }
    else if (d < d1) { d2 = d1; i2 = i1; d1 = d; i1 = j; }
    else if (d < d2) { d2 = d; i2 = j; }
  }
  float r0 = 1.f / (d0 + 1e-8f), r1 = 1.f / (d1 + 1e-8f), r2 = 1.f / (d2 + 1e-8f);
  float s = r0 + r1 + r2;
  float w0 = r0 / s, w1 = r1 / s, w2 = r2 / s;
  float* row = cat + (size_t)t * (C1 + C2) + C1;
  const float* P2 = p2 + (size_t)b * S2 * C2;
  for (int c = 0; c < C2; ++c)
    row[c] = w0 * P2[(size_t)i0 * C2 + c] + w1 * P2[(size_t)i1 * C2 + c]
           + w2 * P2[(size_t)i2 * C2 + c];
}

// mn[b,i] = min_j ||A[b,i]-Bp[b,j]||^2
__global__ void k_nnmin(const float* __restrict__ A, const float* __restrict__ Bp,
                        float* __restrict__ mn, int Na, int Nb, int total)
{
  int t = blockIdx.x * 256 + threadIdx.x;
  if (t >= total) return;
  int b = t / Na;
  const float* P = Bp + (size_t)b * Nb * 3;
  float qx = A[(size_t)t * 3], qy = A[(size_t)t * 3 + 1], qz = A[(size_t)t * 3 + 2];
  float m = 1e30f;
  for (int j = 0; j < Nb; ++j) {
    float dx = P[j * 3] - qx, dy = P[j * 3 + 1] - qy, dz = P[j * 3 + 2] - qz;
    m = fminf(m, dx * dx + dy * dy + dz * dz);
  }
  mn[t] = m;
}

__global__ void k_final(const float* __restrict__ a, const float* __restrict__ b,
                        int n, float* __restrict__ out, float cbr)
{
  __shared__ float s[256];
  float acc = 0.f;
  for (int i = threadIdx.x; i < n; i += 256) acc += a[i] + b[i];
  s[threadIdx.x] = acc;
  __syncthreads();
  for (int o = 128; o > 0; o >>= 1) {
    if (threadIdx.x < o) s[threadIdx.x] += s[threadIdx.x + o];
    __syncthreads();
  }
  if (threadIdx.x == 0) { out[0] = s[0] / (float)n; out[1] = cbr; }
}

// =====================================================================
// Host side
// =====================================================================
struct Lin { const float* w; const float* b; int cin, cout; };
struct TB  { Lin d0, d1, fc1, fc2, g0, g1, wk, wq, wv; int d; };

struct Arena {
  char* base; size_t off; size_t cap;
  float* f(size_t n) { size_t a = (off + 255) & ~(size_t)255; off = a + n * sizeof(float); return (float*)(base + a); }
  int*   i(size_t n) { size_t a = (off + 255) & ~(size_t)255; off = a + n * sizeof(int);   return (int*)(base + a); }
};

static Lin takeLin(void* const* d_in, int& c, int cin, int cout) {
  Lin L;                       // sorted dict keys: 'b' before 'w'
  L.b = (const float*)d_in[c++];
  L.w = (const float*)d_in[c++];
  L.cin = cin; L.cout = cout;
  return L;
}
static TB takeTB(void* const* d_in, int& c, int d) {
  TB t; t.d = d;               // sorted keys: delta, fc1, fc2, gamma, wk, wq, wv
  t.d0  = takeLin(d_in, c, 3, d);
  t.d1  = takeLin(d_in, c, d, d);
  t.fc1 = takeLin(d_in, c, d, d);
  t.fc2 = takeLin(d_in, c, d, d);
  t.g0  = takeLin(d_in, c, d, d);
  t.g1  = takeLin(d_in, c, d, d);
  t.wk  = takeLin(d_in, c, d, d);
  t.wq  = takeLin(d_in, c, d, d);
  t.wv  = takeLin(d_in, c, d, d);
  return t;
}

static void gemm(hipStream_t st, const float* A, const Lin& L,
                 const float* res, float* C, int M, int relu) {
  dim3 g((M + GTM - 1) / GTM, (L.cout + GTN - 1) / GTN);
  k_gemm<<<g, 128, 0, st>>>(A, L.w, L.b, res, C, M, L.cin, L.cout, relu);
}

static void bnrelu(hipStream_t st, float* X, int M, int C, Arena& ar) {
  float* mv = ar.f((size_t)2 * C);
  k_bn_stats<<<C, 256, 0, st>>>(X, M, C, mv);
  int tot = M * C;
  k_bn_apply<<<(tot + 255) / 256, 256, 0, st>>>(X, mv, tot, C, 1);
}

static void set_abs(hipStream_t st, const float* xyz_l, const float* feats,
                    int Npt, int S, float radius, const Lin* Ls, int nL,
                    float* new_xyz, float* new_f, Arena& ar)
{
  size_t save = ar.off;
  int Cin = Ls[0].cin - 3;
  int* fi = ar.i((size_t)BATCH * S);
  k_fps<<<BATCH, 256, 0, st>>>(xyz_l, Npt, S, fi);
  int tot = BATCH * S * 3;
  k_gather<<<(tot + 255) / 256, 256, 0, st>>>(xyz_l, fi, new_xyz, Npt, S, 3, tot);
  int* gi = ar.i((size_t)BATCH * S * NSAMP);
  tot = BATCH * S;
  k_ball<<<(tot + 255) / 256, 256, 0, st>>>(xyz_l, new_xyz, gi, Npt, S, radius * radius, tot);
  int M = BATCH * S * NSAMP;
  float* g = ar.f((size_t)M * (3 + Cin));
  k_group<<<(M + 255) / 256, 256, 0, st>>>(xyz_l, feats, gi, new_xyz, g, Npt, S, Cin, M);
  const float* cur = g;
  for (int l = 0; l < nL; ++l) {
    float* h = ar.f((size_t)M * Ls[l].cout);
    gemm(st, cur, Ls[l], nullptr, h, M, 0);
    bnrelu(st, h, M, Ls[l].cout, ar);
    cur = h;
  }
  int Cout = Ls[nL - 1].cout;
  tot = BATCH * S * Cout;
  k_maxpool<<<(tot + 255) / 256, 256, 0, st>>>(cur, new_f, S, Cout, tot);
  ar.off = save;
}

// Transformer block processed one batch at a time (no BN inside -> legal),
// keeping the (S,32,d) intermediates 4x smaller.
static void transformer(hipStream_t st, const float* xyz_l, float* feats,
                        const TB& p, int S, Arena& ar)
{
  const int d = p.d;
  for (int b = 0; b < BATCH; ++b) {
    size_t save = ar.off;
    const float* X = xyz_l + (size_t)b * S * 3;
    float* F = feats + (size_t)b * S * d;
    int P = S;
    float* x  = ar.f((size_t)P * d); gemm(st, F, p.fc1, nullptr, x,  P, 0);
    float* q  = ar.f((size_t)P * d); gemm(st, x, p.wq,  nullptr, q,  P, 0);
    float* kx = ar.f((size_t)P * d); gemm(st, x, p.wk,  nullptr, kx, P, 0);
    float* vx = ar.f((size_t)P * d); gemm(st, x, p.wv,  nullptr, vx, P, 0);
    int* knn = ar.i((size_t)P * KNNK);
    k_knn<<<(P + 255) / 256, 256, 0, st>>>(X, knn, S, P);
    int Mb = P * KNNK;
    float* pxyz = ar.f((size_t)Mb * 3);
    k_posxyz<<<(Mb + 255) / 256, 256, 0, st>>>(X, knn, pxyz, S, Mb);
    float* posh = ar.f((size_t)Mb * d);
    gemm(st, pxyz, p.d0, nullptr, posh, Mb, 1);
    float* pos = ar.f((size_t)Mb * d);
    gemm(st, posh, p.d1, nullptr, pos, Mb, 0);
    float* tt = posh;  // reuse (posh dead)
    int tot = Mb * d;
    k_build_t<<<(tot + 255) / 256, 256, 0, st>>>(q, kx, knn, pos, tt, S, d, tot);
    float* ah = ar.f((size_t)Mb * d);
    gemm(st, tt, p.g0, nullptr, ah, Mb, 1);
    float* attn = tt;  // reuse (tt consumed by previous gemm)
    gemm(st, ah, p.g1, nullptr, attn, Mb, 0);
    tot = P * d;
    k_softmax<<<(tot + 255) / 256, 256, 0, st>>>(attn, d, rsqrtf((float)d), tot);
    float* rb = x;     // reuse
    k_agg<<<(tot + 255) / 256, 256, 0, st>>>(attn, vx, knn, pos, rb, S, d, tot);
    gemm(st, rb, p.fc2, F, F, P, 0);   // + residual
    ar.off = save;
  }
}

static void feat_prop(hipStream_t st, const float* x1, const float* x2,
                      const float* p1, const float* p2, int N1, int S2,
                      int C1, int C2, const Lin* Ls, float* outP, Arena& ar)
{
  size_t save = ar.off;
  int P = BATCH * N1, Ct = C1 + C2;
  float* cat = ar.f((size_t)P * Ct);
  int tot = P * C1;
  k_copycat<<<(tot + 255) / 256, 256, 0, st>>>(p1, cat, C1, Ct, tot);
  k_interp3<<<(P + 255) / 256, 256, 0, st>>>(x1, x2, p2, cat, N1, S2, C1, C2, P);
  float* h = ar.f((size_t)P * Ls[0].cout);
  gemm(st, cat, Ls[0], nullptr, h, P, 0); bnrelu(st, h, P, Ls[0].cout, ar);
  gemm(st, h, Ls[1], nullptr, outP, P, 0); bnrelu(st, outP, P, Ls[1].cout, ar);
  ar.off = save;
}

extern "C" void kernel_launch(void* const* d_in, const int* in_sizes, int n_in,
                              void* d_out, int out_size, void* d_ws, size_t ws_size,
                              hipStream_t stream)
{
  (void)out_size;
  // Top-level dict may be flattened in insertion (xyz,params,noise) or
  // sorted (noise,params,xyz) order; disambiguate by element count.
  const float* xyz; const float* noise;
  if (in_sizes[0] == BATCH * NPTS * 3) {
    xyz = (const float*)d_in[0]; noise = (const float*)d_in[n_in - 1];
  } else {
    noise = (const float*)d_in[0]; xyz = (const float*)d_in[n_in - 1];
  }
  int c = 1;  // params leaves, sorted-key pytree order
  TB attn1 = takeTB(d_in, c, 128), attn2 = takeTB(d_in, c, 256),
     attn3 = takeTB(d_in, c, 512), attn4 = takeTB(d_in, c, 256);
  TB dattn1 = takeTB(d_in, c, 64),  dattn2 = takeTB(d_in, c, 128),
     dattn3 = takeTB(d_in, c, 256), dattn4 = takeTB(d_in, c, 512);
  Lin emb0 = takeLin(d_in, c, 3, 64),   emb1 = takeLin(d_in, c, 64, 128);
  Lin fc_dec = takeLin(d_in, c, 256, 256), fc_enc = takeLin(d_in, c, 256, 256);
  Lin fp1L[2] = { takeLin(d_in, c, 256, 128), takeLin(d_in, c, 128, 64) };
  Lin fp2L[2] = { takeLin(d_in, c, 384, 256), takeLin(d_in, c, 256, 128) };
  Lin fp3L[2] = { takeLin(d_in, c, 768, 512), takeLin(d_in, c, 512, 256) };
  Lin fp4L[2] = { takeLin(d_in, c, 768, 512), takeLin(d_in, c, 512, 512) };
  Lin proj0 = takeLin(d_in, c, 64, 64), proj1 = takeLin(d_in, c, 64, 3);
  Lin sa1L[3] = { takeLin(d_in, c, 131, 64),  takeLin(d_in, c, 64, 64),   takeLin(d_in, c, 64, 128) };
  Lin sa2L[3] = { takeLin(d_in, c, 131, 128), takeLin(d_in, c, 128, 128), takeLin(d_in, c, 128, 256) };
  Lin sa3L[3] = { takeLin(d_in, c, 259, 256), takeLin(d_in, c, 256, 256), takeLin(d_in, c, 256, 512) };
  Lin sa4L[3] = { takeLin(d_in, c, 515, 512), takeLin(d_in, c, 512, 512), takeLin(d_in, c, 512, 256) };

  Arena ar{ (char*)d_ws, 0, ws_size };
  // Persistent buffers
  float* l1x = ar.f((size_t)BATCH * 1024 * 3);
  float* l2x = ar.f((size_t)BATCH * 512 * 3);
  float* l3x = ar.f((size_t)BATCH * 256 * 3);
  float* l4x = ar.f((size_t)BATCH * 128 * 3);
  float* l0f = ar.f((size_t)BATCH * NPTS * 128);
  float* l1f = ar.f((size_t)BATCH * 1024 * 128);
  float* l2f = ar.f((size_t)BATCH * 512 * 256);
  float* l3f = ar.f((size_t)BATCH * 256 * 512);
  float* l4f = ar.f((size_t)BATCH * 128 * 256);
  float* zv  = ar.f((size_t)BATCH * 256);
  float* f3  = ar.f((size_t)BATCH * 256 * 512);
  float* f2b = ar.f((size_t)BATCH * 512 * 256);
  float* f1b = ar.f((size_t)BATCH * 1024 * 128);
  float* f0b = ar.f((size_t)BATCH * NPTS * 64);
  float* mnA = ar.f((size_t)BATCH * NPTS);
  float* mnB = ar.f((size_t)BATCH * NPTS);

  const int P0 = BATCH * NPTS;
  // ---- embed ----
  { size_t save = ar.off;
    float* h = ar.f((size_t)P0 * 64);
    gemm(stream, xyz, emb0, nullptr, h, P0, 0); bnrelu(stream, h, P0, 64, ar);
    gemm(stream, h, emb1, nullptr, l0f, P0, 0); bnrelu(stream, l0f, P0, 128, ar);
    ar.off = save;
  }
  // ---- encoder ----
  set_abs(stream, xyz, l0f, NPTS, 1024, 0.1f, sa1L, 3, l1x, l1f, ar);
  transformer(stream, l1x, l1f, attn1, 1024, ar);
  set_abs(stream, l1x, l1f, 1024, 512, 0.2f, sa2L, 3, l2x, l2f, ar);
  transformer(stream, l2x, l2f, attn2, 512, ar);
  set_abs(stream, l2x, l2f, 512, 256, 0.4f, sa3L, 3, l3x, l3f, ar);
  transformer(stream, l3x, l3f, attn3, 256, ar);
  set_abs(stream, l3x, l3f, 256, 128, 0.6f, sa4L, 3, l4x, l4f, ar);
  transformer(stream, l4x, l4f, attn4, 128, ar);
  // ---- bottleneck: latent -> power-norm channel -> decode vector ----
  { size_t save = ar.off;
    float* lat = ar.f((size_t)BATCH * 256);
    int tot = BATCH * 256;
    k_mean1<<<(tot + 255) / 256, 256, 0, stream>>>(l4f, lat, 128, 256, tot);
    float* tt = ar.f((size_t)BATCH * 256);
    gemm(stream, lat, fc_enc, nullptr, tt, BATCH, 0);
    float* y = ar.f((size_t)BATCH * 256);
    k_powernoise<<<BATCH, 256, 0, stream>>>(tt, noise, y);
    gemm(stream, y, fc_dec, nullptr, zv, BATCH, 0);
    ar.off = save;
  }
  // ---- fp4 (S2==1 broadcast) ----
  { size_t save = ar.off;
    int P = BATCH * 256, Ct = 768;
    float* cat = ar.f((size_t)P * Ct);
    int tot = P * Ct;
    k_concat_b<<<(tot + 255) / 256, 256, 0, stream>>>(l3f, zv, cat, 512, 256, 256, tot);
    float* h = ar.f((size_t)P * 512);
    gemm(stream, cat, fp4L[0], nullptr, h, P, 0);  bnrelu(stream, h, P, 512, ar);
    gemm(stream, h, fp4L[1], nullptr, f3, P, 0);   bnrelu(stream, f3, P, 512, ar);
    ar.off = save;
  }
  transformer(stream, l3x, f3, dattn4, 256, ar);
  feat_prop(stream, l2x, l3x, l2f, f3, 512, 256, 256, 512, fp3L, f2b, ar);
  transformer(stream, l2x, f2b, dattn3, 512, ar);
  feat_prop(stream, l1x, l2x, l1f, f2b, 1024, 512, 128, 256, fp2L, f1b, ar);
  transformer(stream, l1x, f1b, dattn2, 1024, ar);
  feat_prop(stream, xyz, l1x, l0f, f1b, NPTS, 1024, 128, 128, fp1L, f0b, ar);
  transformer(stream, xyz, f0b, dattn1, NPTS, ar);
  // ---- projection head -> recon straight into d_out ----
  float* recon = (float*)d_out;
  { size_t save = ar.off;
    float* h = ar.f((size_t)P0 * 64);
    gemm(stream, f0b, proj0, nullptr, h, P0, 0); bnrelu(stream, h, P0, 64, ar);
    gemm(stream, h, proj1, nullptr, recon, P0, 0);
  }
  // ---- Chamfer distance + CBR ----
  int tot = BATCH * NPTS;
  k_nnmin<<<(tot + 255) / 256, 256, 0, stream>>>(xyz, recon, mnA, NPTS, NPTS, tot);
  k_nnmin<<<(tot + 255) / 256, 256, 0, stream>>>(recon, xyz, mnB, NPTS, NPTS, tot);
  k_final<<<1, 256, 0, stream>>>(mnA, mnB, tot, recon + (size_t)P0 * 3,
                                 256.f / (3.f * (float)NPTS));
}
